// PointNet2_MSG_large_50732153700752
// MI455X (gfx1250) — compile-verified, matrix operand order checked by reading, not run on hardware
//
#include <hip/hip_runtime.h>
#include <hip/hip_bf16.h>

typedef __attribute__((ext_vector_type(16))) _Float16 v16h;
typedef __attribute__((ext_vector_type(8)))  _Float16 v8h;
typedef __attribute__((ext_vector_type(8)))  float    v8f;

// ---------------- layout ----------------
// points:       pts[b][i][3]                       (f32)
// features:     feat[c][b*n + i]                   (f32, channel-major)
// GEMM input:   Xt[m][cinp]                        (f16, element-major, channels zero-padded to cinp%32==0)
// GEMM weights: Wh[r][cinp], r < coutp16           (f16, zero-padded)
// GEMM output:  Y[c][M]                            (f32)

// xyz (B,3,N) -> pts (B,N,3)
__global__ void k_xyz_to_pts(const float* __restrict__ xyz, float* __restrict__ pts, int B, int n) {
  int t = blockIdx.x * blockDim.x + threadIdx.x;
  int total = B * n * 3;
  if (t >= total) return;
  int d = t % 3; int i = (t / 3) % n; int b = t / (3 * n);
  pts[t] = xyz[((size_t)b * 3 + d) * n + i];
}

// xyz (B,3,N) -> feat0 [3][B*N]
__global__ void k_xyz_to_feat(const float* __restrict__ xyz, float* __restrict__ f, int B, int n) {
  int t = blockIdx.x * blockDim.x + threadIdx.x;
  int total = 3 * B * n;
  if (t >= total) return;
  int c = t / (B * n); int r = t % (B * n); int b = r / n; int i = r % n;
  f[t] = xyz[((size_t)b * 3 + c) * n + i];
}

// ---------------- farthest point sampling ----------------
__global__ void k_fps(const float* __restrict__ pts, float* __restrict__ dist,
                      int* __restrict__ fidx, int n, int S) {
  int b = blockIdx.x; int t = threadIdx.x; int nt = blockDim.x;
  const float* P = pts + (size_t)b * n * 3;
  float* D = dist + (size_t)b * n;
  int* F = fidx + b * S;
  __shared__ float sv[256];
  __shared__ int   si[256];
  __shared__ int   sfar;
  for (int i = t; i < n; i += nt) D[i] = 1e10f;
  if (t == 0) sfar = 0;
  __syncthreads();
  for (int it = 0; it < S; ++it) {
    int far = sfar;
    if (t == 0) F[it] = far;
    float cx = P[far * 3 + 0], cy = P[far * 3 + 1], cz = P[far * 3 + 2];
    float best = -1.0f; int bi = 0;
    for (int i = t; i < n; i += nt) {
      float dx = P[i * 3 + 0] - cx, dy = P[i * 3 + 1] - cy, dz = P[i * 3 + 2] - cz;
      float d = dx * dx + dy * dy + dz * dz;
      float dd = fminf(D[i], d); D[i] = dd;
      if (dd > best) { best = dd; bi = i; }
    }
    sv[t] = best; si[t] = bi;
    __syncthreads();
    for (int s2 = nt >> 1; s2 > 0; s2 >>= 1) {
      if (t < s2) {
        float ov = sv[t + s2]; int oi = si[t + s2];
        if (ov > sv[t] || (ov == sv[t] && oi < si[t])) { sv[t] = ov; si[t] = oi; }
      }
      __syncthreads();
    }
    if (t == 0) sfar = si[0];
    __syncthreads();
  }
}

__global__ void k_gather_pts(const float* __restrict__ pts, const int* __restrict__ fidx,
                             float* __restrict__ outp, int B, int n, int S) {
  int t = blockIdx.x * blockDim.x + threadIdx.x;
  if (t >= B * S) return;
  int b = t / S;
  int id = fidx[t];
  outp[t * 3 + 0] = pts[((size_t)b * n + id) * 3 + 0];
  outp[t * 3 + 1] = pts[((size_t)b * n + id) * 3 + 1];
  outp[t * 3 + 2] = pts[((size_t)b * n + id) * 3 + 2];
}

// ---------------- ball query: one wave32 per (b,s) ----------------
__global__ void k_ballq(const float* __restrict__ pts, const float* __restrict__ nx,
                        int* __restrict__ gi, int n, int S, int K, float r2, int totalWaves) {
  int gw = (blockIdx.x * blockDim.x + threadIdx.x) >> 5;
  if (gw >= totalWaves) return;            // wave-uniform
  int lane = threadIdx.x & 31;
  int b = gw / S, s = gw - b * S;
  const float* P = pts + (size_t)b * n * 3;
  float cx = nx[(b * S + s) * 3 + 0], cy = nx[(b * S + s) * 3 + 1], cz = nx[(b * S + s) * 3 + 2];
  int* out = gi + (size_t)(b * S + s) * K;
  int cnt = 0; int first = -1;
  for (int base = 0; base < n && cnt < K; base += 32) {
    int i = base + lane;
    bool inr = false;
    if (i < n) {
      float dx = P[i * 3 + 0] - cx, dy = P[i * 3 + 1] - cy, dz = P[i * 3 + 2] - cz;
      inr = (dx * dx + dy * dy + dz * dz) <= r2;
    }
    unsigned mask = (unsigned)__ballot(inr);
    if (first < 0 && mask) first = base + __ffs(mask) - 1;
    int pos = cnt + __popc(mask & ((1u << lane) - 1u));
    if (inr && pos < K) out[pos] = i;
    cnt += __popc(mask);
  }
  if (first < 0) first = 0;
  if (cnt > K) cnt = K;
  for (int j = cnt + lane; j < K; j += 32) out[j] = first;
}

// ---------------- grouping -> f16 Xt[m][cinp] ----------------
__global__ void k_group_h(const float* __restrict__ feat, const float* __restrict__ pts,
                          const float* __restrict__ nx, const int* __restrict__ gi,
                          _Float16* __restrict__ Xt, int B, int n, int S, int K, int Cf, int cinp) {
  int M = B * S * K;
  int t = blockIdx.x * blockDim.x + threadIdx.x;
  if (t >= M * cinp) return;
  int m = t / cinp; int c = t % cinp;
  int cin = Cf + 3;
  float v = 0.0f;
  if (c < cin) {
    int b = m / (S * K); int rem = m % (S * K); int s = rem / K; int kk = rem % K;
    int id = gi[((size_t)b * S + s) * K + kk];
    if (c < Cf) v = feat[(size_t)c * ((size_t)B * n) + (size_t)b * n + id];
    else {
      int d = c - Cf;
      v = pts[((size_t)b * n + id) * 3 + d] - nx[(b * S + s) * 3 + d];
    }
  }
  Xt[t] = (_Float16)v;
}

// ---------------- weight pre-pack: f32 [cout][cin] -> f16 [coutp16][cinp], zero padded ----------------
__global__ void k_pack_w(const float* __restrict__ W, _Float16* __restrict__ Wh,
                         int cout, int cin, int coutp, int cinp) {
  int t = blockIdx.x * blockDim.x + threadIdx.x;
  if (t >= coutp * cinp) return;
  int r = t / cinp; int c = t % cinp;
  Wh[t] = (r < cout && c < cin) ? (_Float16)W[(size_t)r * cin + c] : (_Float16)0.0f;
}

// ---------------- WMMA GEMM: Y[cout][M] = Wh @ Xt^T (+bias) ----------------
// Wh: [coutp16][cinp] f16, Xt: [M][cinp] f16. Wave tile: 16 rows x 64 cols.
// Software-pipelined: issue K-step k+1's 10x b128 loads, then 4 WMMAs on step k's operands.
__device__ __forceinline__ v16h ld16h(const _Float16* p) {
  v8h lo = *(const v8h*)p;
  v8h hi = *(const v8h*)(p + 16);
  return __builtin_shufflevector(lo, hi, 0,1,2,3,4,5,6,7,8,9,10,11,12,13,14,15);
}

__global__ void __launch_bounds__(128, 1)
k_gemm(const _Float16* __restrict__ Wh, const float* __restrict__ bias,
       const _Float16* __restrict__ Xt, float* __restrict__ Y,
       int cout, int cinp, int M) {
  int lane = threadIdx.x & 31;
  int wave = blockIdx.x * (blockDim.x >> 5) + (threadIdx.x >> 5);
  int mt = M >> 6;                          // 64-column tiles (all M are multiples of 64)
  int ct = (cout + 15) >> 4;
  if (wave >= mt * ct) return;              // wave-uniform
  int tc = wave / mt;
  int tm = wave - tc * mt;
  int col0 = (tm << 6) + (lane & 15);
  int row0 = tc << 4;
  int r    = row0 + (lane & 15);
  int koff = (lane >> 4) << 3;              // 0 or 8
  const _Float16* wp  = Wh + (size_t)r * cinp + koff;
  const _Float16* xp0 = Xt + (size_t)(col0 +  0) * cinp + koff;
  const _Float16* xp1 = Xt + (size_t)(col0 + 16) * cinp + koff;
  const _Float16* xp2 = Xt + (size_t)(col0 + 32) * cinp + koff;
  const _Float16* xp3 = Xt + (size_t)(col0 + 48) * cinp + koff;
  v8f acc0 = {}, acc1 = {}, acc2 = {}, acc3 = {};
  // prologue: load K-step 0
  v16h a  = ld16h(wp);
  v16h b0 = ld16h(xp0);
  v16h b1 = ld16h(xp1);
  v16h b2 = ld16h(xp2);
  v16h b3 = ld16h(xp3);
  for (int k0 = 32; k0 < cinp; k0 += 32) {
    v16h an  = ld16h(wp  + k0);
    v16h bn0 = ld16h(xp0 + k0);
    v16h bn1 = ld16h(xp1 + k0);
    v16h bn2 = ld16h(xp2 + k0);
    v16h bn3 = ld16h(xp3 + k0);
    acc0 = __builtin_amdgcn_wmma_f32_16x16x32_f16(false, a, false, b0, (short)0, acc0, false, false);
    acc1 = __builtin_amdgcn_wmma_f32_16x16x32_f16(false, a, false, b1, (short)0, acc1, false, false);
    acc2 = __builtin_amdgcn_wmma_f32_16x16x32_f16(false, a, false, b2, (short)0, acc2, false, false);
    acc3 = __builtin_amdgcn_wmma_f32_16x16x32_f16(false, a, false, b3, (short)0, acc3, false, false);
    a = an; b0 = bn0; b1 = bn1; b2 = bn2; b3 = bn3;
  }
  acc0 = __builtin_amdgcn_wmma_f32_16x16x32_f16(false, a, false, b0, (short)0, acc0, false, false);
  acc1 = __builtin_amdgcn_wmma_f32_16x16x32_f16(false, a, false, b1, (short)0, acc1, false, false);
  acc2 = __builtin_amdgcn_wmma_f32_16x16x32_f16(false, a, false, b2, (short)0, acc2, false, false);
  acc3 = __builtin_amdgcn_wmma_f32_16x16x32_f16(false, a, false, b3, (short)0, acc3, false, false);
  int rbase = row0 + koff;                  // D rows: j (+8 for upper half-lanes)
#pragma unroll
  for (int j = 0; j < 8; ++j) {
    int rr = rbase + j;
    if (rr < cout) {
      float bi = bias ? bias[rr] : 0.0f;
      float* yr = Y + (size_t)rr * M;
      yr[col0 +  0] = acc0[j] + bi;
      yr[col0 + 16] = acc1[j] + bi;
      yr[col0 + 32] = acc2[j] + bi;
      yr[col0 + 48] = acc3[j] + bi;
    }
  }
}

// ---------------- batch-norm stats (training mode, over M) ----------------
__global__ void k_bn_stats(const float* __restrict__ Y, const float* __restrict__ g,
                           float* __restrict__ mean, float* __restrict__ inv, int M, float eps) {
  int c = blockIdx.x;
  const float* row = Y + (size_t)c * M;
  float s = 0.0f, sq = 0.0f;
  for (int i = threadIdx.x; i < M; i += blockDim.x) { float v = row[i]; s += v; sq += v * v; }
  __shared__ float ss[256], s2[256];
  ss[threadIdx.x] = s; s2[threadIdx.x] = sq;
  __syncthreads();
  for (int st = 128; st > 0; st >>= 1) {
    if (threadIdx.x < st) { ss[threadIdx.x] += ss[threadIdx.x + st]; s2[threadIdx.x] += s2[threadIdx.x + st]; }
    __syncthreads();
  }
  if (threadIdx.x == 0) {
    float mu = ss[0] / (float)M;
    float var = fmaxf(s2[0] / (float)M - mu * mu, 0.0f);
    mean[c] = mu;
    inv[c] = g[c] * rsqrtf(var + eps);
  }
}

// BN+ReLU -> f16 Xt[m][cinp] (chain continues into another GEMM)
__global__ void k_bn_relu_h(const float* __restrict__ Y, const float* __restrict__ mean,
                            const float* __restrict__ inv, const float* __restrict__ be,
                            _Float16* __restrict__ Xt, int cout, int cinp, int M) {
  int t = blockIdx.x * blockDim.x + threadIdx.x;
  if (t >= M * cinp) return;
  int m = t / cinp; int c = t % cinp;
  float v = 0.0f;
  if (c < cout) {
    float y = Y[(size_t)c * M + m];
    v = (y - mean[c]) * inv[c] + be[c];
    v = v > 0.0f ? v : 0.0f;
  }
  Xt[t] = (_Float16)v;
}

// BN+ReLU -> f32 [c][M] (may run in place on Y)
__global__ void k_bn_relu_f(const float* Y, const float* __restrict__ mean,
                            const float* __restrict__ inv, const float* __restrict__ be,
                            float* X, int cout, int M) {
  int t = blockIdx.x * blockDim.x + threadIdx.x;
  if (t >= cout * M) return;
  int c = t / M;
  float v = (Y[t] - mean[c]) * inv[c] + be[c];
  X[t] = v > 0.0f ? v : 0.0f;
}

// ---------------- max pool over nsample ----------------
__global__ void k_maxpool(const float* __restrict__ Y, float* __restrict__ feat,
                          int B, int S, int K, int cout, int cofs) {
  int t = blockIdx.x * blockDim.x + threadIdx.x;
  if (t >= cout * B * S) return;
  int c = t / (B * S); int r = t % (B * S);
  const float* p = Y + (size_t)c * ((size_t)B * S * K) + (size_t)r * K;
  float m = p[0];
  for (int kk = 1; kk < K; ++kk) m = fmaxf(m, p[kk]);
  feat[(size_t)(cofs + c) * ((size_t)B * S) + r] = m;
}

// ---------------- 3-NN + inverse-distance weights ----------------
__global__ void k_knn3(const float* __restrict__ pts1, const float* __restrict__ pts2,
                       int* __restrict__ idx3, float* __restrict__ w3, int B, int n1, int n2) {
  int t = blockIdx.x * blockDim.x + threadIdx.x;
  if (t >= B * n1) return;
  int b = t / n1;
  const float* P2 = pts2 + (size_t)b * n2 * 3;
  float x = pts1[t * 3 + 0], y = pts1[t * 3 + 1], z = pts1[t * 3 + 2];
  float d0 = 3.4e38f, d1 = 3.4e38f, d2 = 3.4e38f;
  int i0 = 0, i1 = 0, i2 = 0;
  for (int j = 0; j < n2; ++j) {
    float dx = P2[j * 3 + 0] - x, dy = P2[j * 3 + 1] - y, dz = P2[j * 3 + 2] - z;
    float d = dx * dx + dy * dy + dz * dz;
    if (d < d0)      { d2 = d1; i2 = i1; d1 = d0; i1 = i0; d0 = d; i0 = j; }
    else if (d < d1) { d2 = d1; i2 = i1; d1 = d;  i1 = j; }
    else if (d < d2) { d2 = d;  i2 = j; }
  }
  float r0 = 1.0f / (d0 + 1e-8f), r1 = 1.0f / (d1 + 1e-8f), r2 = 1.0f / (d2 + 1e-8f);
  float s = r0 + r1 + r2;
  idx3[t * 3 + 0] = i0; idx3[t * 3 + 1] = i1; idx3[t * 3 + 2] = i2;
  w3[t * 3 + 0] = r0 / s; w3[t * 3 + 1] = r1 / s; w3[t * 3 + 2] = r2 / s;
}

// ---------------- FP input build -> f16 Xt[m][cinp]: [feat1 | interp(feat2)] ----------------
__global__ void k_fp_build_h(const float* __restrict__ feat1, const float* __restrict__ feat2,
                             const int* __restrict__ idx3, const float* __restrict__ w3,
                             _Float16* __restrict__ Xt, int B, int n1, int n2,
                             int c1, int c2, int cinp) {
  int M = B * n1;
  int t = blockIdx.x * blockDim.x + threadIdx.x;
  if (t >= M * cinp) return;
  int m = t / cinp; int c = t % cinp;
  float v = 0.0f;
  if (c < c1) {
    v = feat1[(size_t)c * M + m];
  } else if (c < c1 + c2) {
    int cc = c - c1; int b = m / n1;
    const float* f2 = feat2 + (size_t)cc * ((size_t)B * n2) + (size_t)b * n2;
    int base = m * 3;
    v = f2[idx3[base]] * w3[base] + f2[idx3[base + 1]] * w3[base + 1] + f2[idx3[base + 2]] * w3[base + 2];
  }
  Xt[t] = (_Float16)v;
}

// Y[C][B*n] -> out[(b*n+i)*C + c]
__global__ void k_out_transpose(const float* __restrict__ Y, float* __restrict__ out,
                                int B, int n, int C) {
  int t = blockIdx.x * blockDim.x + threadIdx.x;
  if (t >= B * n * C) return;
  int c = t % C; int bi = t / C;
  out[t] = Y[(size_t)c * ((size_t)B * n) + bi];
}

// =====================================================================
extern "C" void kernel_launch(void* const* d_in, const int* in_sizes, int n_in,
                              void* d_out, int out_size, void* d_ws, size_t ws_size,
                              hipStream_t stream) {
  const int B = 8, N = 16384;

  // xyz may be first (insertion order) or last (sorted pytree order)
  int xyzIdx, cur;
  if (n_in > 0 && in_sizes[0] == B * 3 * N) { xyzIdx = 0; cur = 1; }
  else { xyzIdx = n_in - 1; cur = 0; }
  const float* xyz = (const float*)d_in[xyzIdx];

  struct Layer { const float *w, *b, *be, *g; int cin, cout; };
  auto takeLayer = [&](int cin, int cout) -> Layer {
    Layer L; L.cin = cin; L.cout = cout; L.w = nullptr;
    const float* o[3]; int no = 0;
    for (int j = 0; j < 4; ++j) {
      const float* p = (const float*)d_in[cur + j];
      if (!L.w && in_sizes[cur + j] == cin * cout) L.w = p;
      else if (no < 3) o[no++] = p;
    }
    L.b = o[0]; L.be = o[1]; L.g = o[2];
    cur += 4; return L;
  };
  struct Fin { const float *w, *b; int cout; };
  auto takeFin = [&](int cout) -> Fin {
    Fin F; F.cout = cout; F.w = nullptr; F.b = nullptr;
    for (int j = 0; j < 2; ++j) {
      const float* p = (const float*)d_in[cur + j];
      if (in_sizes[cur + j] == cout * 128) F.w = p; else F.b = p;
    }
    cur += 2; return F;
  };

  // params pytree order (dict keys sorted): conv1, finals, fp1..fp4, sa1..sa4
  Layer conv1 = takeLayer(128, 128);
  Fin fin0 = takeFin(63), fin1 = takeFin(21), fin2 = takeFin(21);
  Layer fp1L[3] = { takeLayer(128, 128), takeLayer(128, 128), takeLayer(128, 128) };
  Layer fp2L[2] = { takeLayer(352, 256), takeLayer(256, 128) };
  Layer fp3L[2] = { takeLayer(512, 256), takeLayer(256, 256) };
  Layer fp4L[2] = { takeLayer(1536, 256), takeLayer(256, 256) };
  Layer sa1a[3] = { takeLayer(6, 16), takeLayer(16, 16), takeLayer(16, 32) };
  Layer sa1b[3] = { takeLayer(6, 32), takeLayer(32, 32), takeLayer(32, 64) };
  Layer sa2a[3] = { takeLayer(99, 64), takeLayer(64, 64), takeLayer(64, 128) };
  Layer sa2b[3] = { takeLayer(99, 64), takeLayer(64, 96), takeLayer(96, 128) };
  Layer sa3a[3] = { takeLayer(259, 128), takeLayer(128, 196), takeLayer(196, 256) };
  Layer sa3b[3] = { takeLayer(259, 128), takeLayer(128, 196), takeLayer(196, 256) };
  Layer sa4a[3] = { takeLayer(515, 256), takeLayer(256, 256), takeLayer(256, 512) };
  Layer sa4b[3] = { takeLayer(515, 256), takeLayer(256, 384), takeLayer(384, 512) };

  // ---- workspace carve ----
  size_t off = 0;
  auto alloc = [&](size_t nbytes) -> void* {
    void* p = (char*)d_ws + off;
    off = (off + nbytes + 255) & ~(size_t)255;
    return p;
  };
  float* pts0 = (float*)d_out;                               // (B,N,3) = output #1
  float* feat0 = (float*)alloc(sizeof(float) * 3 * B * N);
  float* pts1 = (float*)alloc(sizeof(float) * B * 512 * 3);
  float* pts2 = (float*)alloc(sizeof(float) * B * 256 * 3);
  float* pts3 = (float*)alloc(sizeof(float) * B * 64 * 3);
  float* pts4 = (float*)alloc(sizeof(float) * B * 16 * 3);
  float* feat1 = (float*)alloc(sizeof(float) * 96 * B * 512);
  float* feat2 = (float*)alloc(sizeof(float) * 256 * B * 256);
  float* feat3 = (float*)alloc(sizeof(float) * 512 * B * 64);
  float* feat4 = (float*)alloc(sizeof(float) * 1024 * B * 16);
  float* fpo3 = (float*)alloc(sizeof(float) * 256 * B * 64);
  float* fpo2 = (float*)alloc(sizeof(float) * 256 * B * 256);
  float* fpo1 = (float*)alloc(sizeof(float) * 128 * B * 512);
  float* dist = (float*)alloc(sizeof(float) * B * N);
  int* fidx = (int*)alloc(sizeof(int) * B * 512);
  int* gi = (int*)alloc(sizeof(int) * B * 512 * 32);
  int* idx3 = (int*)alloc(sizeof(int) * (size_t)B * N * 3);
  float* w3 = (float*)alloc(sizeof(float) * (size_t)B * N * 3);
  float* meanv = (float*)alloc(sizeof(float) * 1024);
  float* invv = (float*)alloc(sizeof(float) * 1024);
  _Float16* wpack = (_Float16*)alloc(sizeof(_Float16) * (size_t)524288);
  _Float16* bufXt = (_Float16*)alloc(sizeof(_Float16) * (size_t)16777216);
  float* bufY = (float*)alloc(sizeof(float) * (size_t)16777216);
  (void)ws_size; (void)out_size;

  auto padK = [](int c) { return (c + 31) & ~31; };
  auto padR = [](int c) { return (c + 15) & ~15; };

  auto gemm = [&](const _Float16* Wh, const float* bias, const _Float16* Xt, float* Y,
                  int cout, int cinp, int M) {
    int mt = M >> 6, ct = (cout + 15) >> 4;   // 16x64 tiles per wave
    int waves = mt * ct;
    int blk = (waves + 3) / 4;
    k_gemm<<<blk, 128, 0, stream>>>(Wh, bias, Xt, Y, cout, cinp, M);
  };
  auto packW = [&](const Layer& L) {
    int cinp = padK(L.cin), coutp = padR(L.cout);
    int tot = coutp * cinp;
    k_pack_w<<<(tot + 255) / 256, 256, 0, stream>>>(L.w, wpack, L.cout, L.cin, coutp, cinp);
    return cinp;
  };
  // mode: outF32==nullptr -> f16 Xt (chain continues); else f32 [c][M] into outF32 (may be bufY)
  auto cbr = [&](const Layer& L, int M, float* outF32) {
    int cinp = packW(L);
    gemm(wpack, nullptr, bufXt, bufY, L.cout, cinp, M);
    k_bn_stats<<<L.cout, 256, 0, stream>>>(bufY, L.g, meanv, invv, M, 1e-5f);
    if (outF32) {
      int tot = L.cout * M;
      k_bn_relu_f<<<(tot + 255) / 256, 256, 0, stream>>>(bufY, meanv, invv, L.be, outF32, L.cout, M);
    } else {
      int cop = padK(L.cout);
      int tot = M * cop;
      k_bn_relu_h<<<(tot + 255) / 256, 256, 0, stream>>>(bufY, meanv, invv, L.be, bufXt, L.cout, cop, M);
    }
  };

  auto sa = [&](const float* ptsIn, const float* featIn, int n, int Cf, int S,
                float r0, int K0, const Layer* br0,
                float r1, int K1, const Layer* br1,
                float* ptsOut, float* featOut) {
    k_fps<<<B, 256, 0, stream>>>(ptsIn, dist, fidx, n, S);
    k_gather_pts<<<(B * S + 255) / 256, 256, 0, stream>>>(ptsIn, fidx, ptsOut, B, n, S);
    int cofs = 0;
    const Layer* brs[2] = { br0, br1 };
    float rads[2] = { r0, r1 };
    int Ks[2] = { K0, K1 };
    for (int branch = 0; branch < 2; ++branch) {
      const Layer* Ls = brs[branch];
      int K = Ks[branch];
      float rr = rads[branch];
      int waves = B * S;
      k_ballq<<<(waves * 32 + 255) / 256, 256, 0, stream>>>(ptsIn, ptsOut, gi, n, S, K, rr * rr, waves);
      int M = B * S * K;
      int cinp = padK(Cf + 3);
      int tot = M * cinp;
      k_group_h<<<(tot + 255) / 256, 256, 0, stream>>>(featIn, ptsIn, ptsOut, gi, bufXt, B, n, S, K, Cf, cinp);
      cbr(Ls[0], M, nullptr);
      cbr(Ls[1], M, nullptr);
      cbr(Ls[2], M, bufY);                          // in-place f32 activations for pooling
      int cO = Ls[2].cout;
      k_maxpool<<<(cO * B * S + 255) / 256, 256, 0, stream>>>(bufY, featOut, B, S, K, cO, cofs);
      cofs += cO;
    }
  };

  auto fp = [&](const float* pts1p, const float* pts2p, int n1, int n2,
                const float* feat1p, int c1, const float* feat2p, int c2,
                const Layer* Ls, int nL, float* outFeat /* null -> leave f16 in bufXt */) {
    k_knn3<<<(B * n1 + 255) / 256, 256, 0, stream>>>(pts1p, pts2p, idx3, w3, B, n1, n2);
    int M = B * n1;
    int cinp = padK(c1 + c2);
    int tot = M * cinp;
    k_fp_build_h<<<(tot + 255) / 256, 256, 0, stream>>>(feat1p, feat2p, idx3, w3, bufXt, B, n1, n2, c1, c2, cinp);
    for (int l = 0; l < nL; ++l) {
      float* o = (l == nL - 1) ? outFeat : nullptr;
      cbr(Ls[l], M, o);
    }
  };

  // ---- pipeline ----
  k_xyz_to_pts<<<(B * N * 3 + 255) / 256, 256, 0, stream>>>(xyz, pts0, B, N);
  k_xyz_to_feat<<<(3 * B * N + 255) / 256, 256, 0, stream>>>(xyz, feat0, B, N);

  sa(pts0, feat0, N,   3,   512, 0.05f, 16, sa1a, 0.1f, 32, sa1b, pts1, feat1);   // -> 96ch
  sa(pts1, feat1, 512, 96,  256, 0.1f,  16, sa2a, 0.2f, 32, sa2b, pts2, feat2);   // -> 256ch
  sa(pts2, feat2, 256, 256, 64,  0.2f,  16, sa3a, 0.4f, 32, sa3b, pts3, feat3);   // -> 512ch
  sa(pts3, feat3, 64,  512, 16,  0.4f,  16, sa4a, 0.8f, 32, sa4b, pts4, feat4);   // -> 1024ch

  fp(pts3, pts4, 64,  16,  feat3, 512, feat4, 1024, fp4L, 2, fpo3);
  fp(pts2, pts3, 256, 64,  feat2, 256, fpo3,  256,  fp3L, 2, fpo2);
  fp(pts1, pts2, 512, 256, feat1, 96,  fpo2,  256,  fp2L, 2, fpo1);
  fp(pts0, pts1, N,   512, nullptr, 0, fpo1,  128,  fp1L, 3, nullptr);  // f16 result in bufXt

  {
    int M = B * N;
    cbr(conv1, M, nullptr);                                   // f16 feat in bufXt
    // finals (bias, no BN/ReLU) into packed rows of bufY
    Fin fins[3] = { fin0, fin1, fin2 };
    int rowoff[3] = { 0, 63, 84 };
    for (int f = 0; f < 3; ++f) {
      int cinp = 128, coutp = padR(fins[f].cout);
      int tot = coutp * cinp;
      k_pack_w<<<(tot + 255) / 256, 256, 0, stream>>>(fins[f].w, wpack, fins[f].cout, 128, coutp, cinp);
      gemm(wpack, fins[f].b, bufXt, bufY + (size_t)rowoff[f] * M, fins[f].cout, cinp, M);
    }
    float* out2 = (float*)d_out + (size_t)B * N * 3;
    k_out_transpose<<<(B * N * 105 + 255) / 256, 256, 0, stream>>>(bufY, out2, B, N, 105);
  }
}
